// Head_65266323030687
// MI455X (gfx1250) — matrix operand order repaired and checked by reading, not compiled, not run on hardware
//
#include <hip/hip_runtime.h>
#include <hip/hip_bf16.h>

// ---------------------------------------------------------------------------
// Memorizing-transformer head for MI455X (gfx1250, wave32, WMMA bf16 path).
// ---------------------------------------------------------------------------

typedef __attribute__((ext_vector_type(16))) __bf16 v16bf;
typedef __attribute__((ext_vector_type(8)))  __bf16 v8bf;
typedef __attribute__((ext_vector_type(8)))  float  v8f;

#define WMMA_BF16(A, B, Cacc) \
  __builtin_amdgcn_wmma_f32_16x16x32_bf16(false, (A), false, (B), (short)0, (Cacc), false, false)

#define INF __builtin_inff()

constexpr int   Bb    = 4;
constexpr int   Tt    = 2048;
constexpr int   Cc    = 1024;
constexpr int   Dd    = 64;
constexpr int   Nn    = 16384;
constexpr int   TOPK  = 32;
constexpr int   Mrows = Bb * Tt;          // 8192
constexpr float SCALE = 0.03125f;         // 1/sqrt(1024)

// ---------------------------------------------------------------------------
// Fragment builders (contiguous k = 16*(lane>>4) + e per lane, wave32).
// All hot-path fragments are two 16B contiguous loads per lane (b128).
// ---------------------------------------------------------------------------

// A fragment (16x32, 16-bit): lane m = lane&15; lanes 16-31 take K+16.
// Per lane K spans [kh, kh+15] contiguous.
__device__ inline v16bf frag_a16(const __bf16* tile, int ld, int k0, int lane) {
  int m  = lane & 15;
  int kh = ((lane >> 4) & 1) << 4;
  const __bf16* rp = tile + (size_t)m * ld + k0;
  v8bf lo = *(const v8bf*)(rp + kh);
  v8bf hi = *(const v8bf*)(rp + kh + 8);
  return __builtin_shufflevector(lo, hi, 0,1,2,3,4,5,6,7,8,9,10,11,12,13,14,15);
}

// B fragment (32x16, 16-bit) where B(k,n) comes from a row-major [n][k] array
// (i.e. multiply by the stored matrix transposed). Lane n = lane&15,
// K base = 16*(lane>>4); element e holds K = kb + e -> 16 contiguous bf16.
__device__ inline v16bf frag_bt16(const __bf16* tile, int ld, int k0, int lane) {
  int n  = lane & 15;
  int kb = ((lane >> 4) & 1) << 4;
  const __bf16* rp = tile + (size_t)n * ld + k0 + kb;
  v8bf lo = *(const v8bf*)(rp);
  v8bf hi = *(const v8bf*)(rp + 8);
  return __builtin_shufflevector(lo, hi, 0,1,2,3,4,5,6,7,8,9,10,11,12,13,14,15);
}

// Half-wave (16-lane) reductions: C-frag rows live in one half-wave each.
__device__ inline float halfmax(float v) {
#pragma unroll
  for (int m = 1; m < 16; m <<= 1) v = fmaxf(v, __shfl_xor(v, m, 32));
  return v;
}
__device__ inline float halfsum(float v) {
#pragma unroll
  for (int m = 1; m < 16; m <<= 1) v += __shfl_xor(v, m, 32);
  return v;
}

// ---------------------------------------------------------------------------
// Kernels
// ---------------------------------------------------------------------------

__global__ void k_cvt_bf16(const float* __restrict__ s, __bf16* __restrict__ d, int n) {
  for (int i = blockIdx.x * blockDim.x + threadIdx.x; i < n; i += gridDim.x * blockDim.x)
    d[i] = (__bf16)s[i];
}

// Pack Wq|Wk|Wv transposed: wt[n][c], n in [0,192), c in [0,1024).
__global__ void k_pack_w(const float* __restrict__ Wq, const float* __restrict__ Wk,
                         const float* __restrict__ Wv, __bf16* __restrict__ wt) {
  int i = blockIdx.x * blockDim.x + threadIdx.x;   // i = n*1024 + c
  if (i >= 192 * Cc) return;
  int n = i / Cc, c = i % Cc;
  float v = (n < 64) ? Wq[c * 64 + n]
          : (n < 128) ? Wk[c * 64 + (n - 64)]
                      : Wv[c * 64 + (n - 128)];
  wt[i] = (__bf16)v;
}

// QKV projection: [8192 x 1024] x [1024 x 192].
// q,k stored bf16 [B*T, 64]; v stored transposed bf16 [B][64][T] for PV frags.
// One 16x16 output tile per wave; 8 waves per block.
__global__ void __launch_bounds__(256) k_qkv(
    const __bf16* __restrict__ xb, const __bf16* __restrict__ wt,
    const float* __restrict__ bq, const float* __restrict__ bk,
    const float* __restrict__ bvp,
    __bf16* __restrict__ qb, __bf16* __restrict__ kbuf, __bf16* __restrict__ vt) {
  int lane = threadIdx.x & 31;
  int wave = threadIdx.x >> 5;
  int tile = blockIdx.x * 8 + wave;          // 6144 tiles total
  int mt = tile / 12, nt = tile % 12;
  const __bf16* xt    = xb + (size_t)mt * 16 * Cc;
  const __bf16* wtile = wt + (size_t)nt * 16 * Cc;

  v8f acc = {};
  for (int k0 = 0; k0 < Cc; k0 += 32) {
    if (k0 + 64 < Cc) {
      __builtin_prefetch(xt    + (size_t)(lane & 15) * Cc + k0 + 64, 0, 3);
      __builtin_prefetch(wtile + (size_t)(lane & 15) * Cc + k0 + 64, 0, 3);
    }
    v16bf a = frag_a16(xt, Cc, k0, lane);
    v16bf b = frag_bt16(wtile, Cc, k0, lane);
    acc = WMMA_BF16(a, b, acc);
  }

  int nn = lane & 15, mh = (lane >> 4) << 3;
  int ng = nt * 16 + nn;
#pragma unroll
  for (int r = 0; r < 8; ++r) {
    int row = mt * 16 + mh + r;
    if (ng < 64) {
      qb[(size_t)row * Dd + ng] = (__bf16)(acc[r] + bq[ng]);
    } else if (ng < 128) {
      int col = ng - 64;
      kbuf[(size_t)row * Dd + col] = (__bf16)(acc[r] + bk[col]);
    } else {
      int col = ng - 128;
      int b = row >> 11, t = row & 2047;               // Tt == 2048
      vt[((size_t)b * Dd + col) * Tt + t] = (__bf16)(acc[r] + bvp[col]);
    }
  }
}

// Causal flash attention: one wave per 16-query tile, 32 keys per step.
// S: 4 WMMAs (2 key-subtiles x 2 K-chunks); online softmax with half-wave
// reductions; P (16x32) through LDS; PV: 4 full-K WMMAs against transposed V.
__global__ void __launch_bounds__(32) k_attn(
    const __bf16* __restrict__ qb, const __bf16* __restrict__ kbuf,
    const __bf16* __restrict__ vt, float* __restrict__ att) {
  __shared__ __bf16 lds_p[16 * 32];
  int lane = threadIdx.x & 31;
  int qt = blockIdx.x;
  int b  = blockIdx.y;
  int tq0 = qt * 16;
  const __bf16* qtile = qb   + ((size_t)b * Tt + tq0) * Dd;
  const __bf16* kbase = kbuf + (size_t)b * Tt * Dd;
  const __bf16* vtb   = vt   + (size_t)b * Dd * Tt;    // [64][2048]

  v16bf aq0 = frag_a16(qtile, Dd, 0, lane);
  v16bf aq1 = frag_a16(qtile, Dd, 32, lane);
  v8f o0 = {}, o1 = {}, o2 = {}, o3 = {};
  float l[8], mx[8];
#pragma unroll
  for (int r = 0; r < 8; ++r) { l[r] = 0.0f; mx[r] = -INF; }
  int nn = lane & 15, mh = (lane >> 4) << 3;

  // Process 32 keys per iteration; the second 16-key subtile past the diagonal
  // is fully masked (its key range stays < T, so no OOB reads).
  for (int kt = 0; kt <= qt; kt += 2) {
    int key0 = kt * 16;
    const __bf16* kt0 = kbase + (size_t)key0 * Dd;
    const __bf16* kt1 = kbase + (size_t)(key0 + 16) * Dd;
    v8f s0 = {}, s1 = {};
    s0 = WMMA_BF16(aq0, frag_bt16(kt0, Dd, 0, lane), s0);
    s0 = WMMA_BF16(aq1, frag_bt16(kt0, Dd, 32, lane), s0);
    s1 = WMMA_BF16(aq0, frag_bt16(kt1, Dd, 0, lane), s1);
    s1 = WMMA_BF16(aq1, frag_bt16(kt1, Dd, 32, lane), s1);
#pragma unroll
    for (int r = 0; r < 8; ++r) {
      int m = mh + r;
      int tq = tq0 + m;
      float sv0 = s0[r] * SCALE;
      float sv1 = s1[r] * SCALE;
      if (key0 + nn > tq)      sv0 = -INF;      // causal mask
      if (key0 + 16 + nn > tq) sv1 = -INF;
      float rmax = halfmax(fmaxf(sv0, sv1));
      float nm   = fmaxf(mx[r], rmax);
      float corr = __expf(mx[r] - nm);
      float p0   = __expf(sv0 - nm);
      float p1   = __expf(sv1 - nm);
      l[r]  = l[r] * corr + halfsum(p0 + p1);
      mx[r] = nm;
      o0[r] *= corr; o1[r] *= corr; o2[r] *= corr; o3[r] *= corr;
      lds_p[m * 32 + nn]      = (__bf16)p0;     // C-layout -> LDS (16x32)
      lds_p[m * 32 + 16 + nn] = (__bf16)p1;
    }
    asm volatile("s_wait_dscnt 0" ::: "memory");      // LDS write->read fence
    v16bf ap = frag_a16(lds_p, 32, 0, lane);          // full K=32 A-frag (ds_load_b128)
    o0 = WMMA_BF16(ap, frag_bt16(vtb + (size_t)0  * Tt, Tt, key0, lane), o0);
    o1 = WMMA_BF16(ap, frag_bt16(vtb + (size_t)16 * Tt, Tt, key0, lane), o1);
    o2 = WMMA_BF16(ap, frag_bt16(vtb + (size_t)32 * Tt, Tt, key0, lane), o2);
    o3 = WMMA_BF16(ap, frag_bt16(vtb + (size_t)48 * Tt, Tt, key0, lane), o3);
  }
#pragma unroll
  for (int r = 0; r < 8; ++r) {
    float inv = 1.0f / l[r];
    size_t row = ((size_t)b * Tt + tq0 + mh + r) * Dd;
    att[row + 0  + nn] = o0[r] * inv;
    att[row + 16 + nn] = o1[r] * inv;
    att[row + 32 + nn] = o2[r] * inv;
    att[row + 48 + nn] = o3[r] * inv;
  }
}

// kNN memory attention: 4 waves per 16-query tile. Streams 256-key chunks of
// q @ mem_keys^T score tiles (WMMA) into LDS, per-query top-32 in registers,
// then softmax + f32 mem_vals gather + gated combine with attention output.
__global__ void __launch_bounds__(128) k_knn(
    const __bf16* __restrict__ qb, const __bf16* __restrict__ mkb,
    const float* __restrict__ mem_vals, const float* __restrict__ att,
    const float* __restrict__ gate, float* __restrict__ out) {
  __shared__ float sc[16 * 256];
  __shared__ float wgt[16 * TOPK];
  __shared__ int   sidx[16 * TOPK];
  int tid  = threadIdx.x;
  int lane = tid & 31, wave = tid >> 5;
  int qt = blockIdx.x, b = blockIdx.y;
  int tq0 = qt * 16;
  const __bf16* qtile  = qb  + ((size_t)b * Tt + tq0) * Dd;
  const __bf16* mkbase = mkb + (size_t)b * Nn * Dd;

  v16bf aq0 = frag_a16(qtile, Dd, 0, lane);
  v16bf aq1 = frag_a16(qtile, Dd, 32, lane);
  int nn = lane & 15, mh = (lane >> 4) << 3;

  float tv[TOPK]; int ti[TOPK];
#pragma unroll
  for (int j = 0; j < TOPK; ++j) { tv[j] = -INF; ti[j] = 0; }
  float curmin = -INF; int minpos = 0;

  for (int ck = 0; ck < Nn / 256; ++ck) {
    int key0 = ck * 256;
#pragma unroll
    for (int t = 0; t < 4; ++t) {
      int col0 = (wave * 4 + t) * 16;
      const __bf16* ktile = mkbase + (size_t)(key0 + col0) * Dd;
      v8f s = {};
      s = WMMA_BF16(aq0, frag_bt16(ktile, Dd, 0, lane), s);
      s = WMMA_BF16(aq1, frag_bt16(ktile, Dd, 32, lane), s);
#pragma unroll
      for (int r = 0; r < 8; ++r)
        sc[(mh + r) * 256 + col0 + nn] = s[r] * SCALE;
    }
    __syncthreads();
    if (tid < 16) {                                // one thread per query row
      for (int kk = 0; kk < 256; ++kk) {
        float v = sc[tid * 256 + kk];
        if (v > curmin) {
          tv[minpos] = v; ti[minpos] = key0 + kk;
          curmin = tv[0]; minpos = 0;
#pragma unroll
          for (int j = 1; j < TOPK; ++j)
            if (tv[j] < curmin) { curmin = tv[j]; minpos = j; }
        }
      }
    }
    __syncthreads();
  }

  if (tid < 16) {                                  // softmax over top-32
    float wm = -INF;
#pragma unroll
    for (int j = 0; j < TOPK; ++j) wm = fmaxf(wm, tv[j]);
    float sum = 0.0f;
#pragma unroll
    for (int j = 0; j < TOPK; ++j) { float e = __expf(tv[j] - wm); wgt[tid * TOPK + j] = e; sum += e; }
    float inv = 1.0f / sum;
#pragma unroll
    for (int j = 0; j < TOPK; ++j) { wgt[tid * TOPK + j] *= inv; sidx[tid * TOPK + j] = ti[j]; }
  }
  __syncthreads();

  float g = gate[0];
  const float* mvb = mem_vals + (size_t)b * Nn * Dd;
  for (int item = tid; item < 16 * Dd; item += 128) {
    int m = item >> 6, d = item & 63;
    float acc = 0.0f;
#pragma unroll 4
    for (int j = 0; j < TOPK; ++j)
      acc += wgt[m * TOPK + j] * mvb[(size_t)sidx[m * TOPK + j] * Dd + d];
    size_t o = ((size_t)b * Tt + tq0 + m) * Dd + d;
    out[o] = att[o] * g + acc * (1.0f - g);
  }
}

// ---------------------------------------------------------------------------
// Launch
// ---------------------------------------------------------------------------

extern "C" void kernel_launch(void* const* d_in, const int* in_sizes, int n_in,
                              void* d_out, int out_size, void* d_ws, size_t ws_size,
                              hipStream_t stream) {
  const float* x        = (const float*)d_in[0];
  const float* mem_keys = (const float*)d_in[1];
  const float* mem_vals = (const float*)d_in[2];
  const float* Wq       = (const float*)d_in[3];
  const float* bq       = (const float*)d_in[4];
  const float* Wk       = (const float*)d_in[5];
  const float* bk       = (const float*)d_in[6];
  const float* Wv       = (const float*)d_in[7];
  const float* bv       = (const float*)d_in[8];
  const float* gate     = (const float*)d_in[9];
  float* out = (float*)d_out;

  char* ws = (char*)d_ws;
  size_t off = 0;
  auto alloc = [&](size_t bytes) -> char* {
    char* p = ws + off;
    off = (off + bytes + 255) & ~(size_t)255;
    return p;
  };
  __bf16* xb   = (__bf16*)alloc((size_t)Mrows * Cc * sizeof(__bf16));   // 16 MB
  __bf16* wt   = (__bf16*)alloc((size_t)192 * Cc * sizeof(__bf16));     // 384 KB (transposed)
  __bf16* qb   = (__bf16*)alloc((size_t)Mrows * Dd * sizeof(__bf16));   // 1 MB
  __bf16* kbuf = (__bf16*)alloc((size_t)Mrows * Dd * sizeof(__bf16));   // 1 MB
  __bf16* vt   = (__bf16*)alloc((size_t)Mrows * Dd * sizeof(__bf16));   // 1 MB  [B][64][T]
  __bf16* mkb  = (__bf16*)alloc((size_t)Bb * Nn * Dd * sizeof(__bf16)); // 8 MB
  float*  attb = (float*) alloc((size_t)Mrows * Dd * sizeof(float));    // 2 MB

  k_cvt_bf16<<<2048, 256, 0, stream>>>(x, xb, Mrows * Cc);
  k_pack_w<<<(192 * Cc + 255) / 256, 256, 0, stream>>>(Wq, Wk, Wv, wt);
  k_cvt_bf16<<<2048, 256, 0, stream>>>(mem_keys, mkb, Bb * Nn * Dd);

  // 512 M-tiles x 12 N-tiles = 6144 wave-tiles, 8 waves/block
  k_qkv<<<768, 256, 0, stream>>>(xb, wt, bq, bk, bv, qb, kbuf, vt);

  k_attn<<<dim3(Tt / 16, Bb), 32, 0, stream>>>(qb, kbuf, vt, attb);

  k_knn<<<dim3(Tt / 16, Bb), 128, 0, stream>>>(qb, mkb, mem_vals, attb, gate, out);
}
